// LinearAttention_43619687858932
// MI455X (gfx1250) — matrix operand order repaired and checked by reading, not compiled, not run on hardware
//
#include <hip/hip_runtime.h>
#include <hip/hip_bf16.h>

typedef __attribute__((ext_vector_type(16))) _Float16 v16h;
typedef __attribute__((ext_vector_type(8)))  _Float16 v8h;
typedef __attribute__((ext_vector_type(4)))  _Float16 v4h;
typedef __attribute__((ext_vector_type(8)))  float    v8f;
typedef __attribute__((ext_vector_type(4)))  float    v4f;

#define N_   4
#define L_   4096
#define S_   4096
#define H_   8
#define D_   64
#define EPSC 1e-6f
#define CH   256      // s-rows per workgroup, pass 1
#define LTILE 128     // l-rows per workgroup, pass 2

__device__ __forceinline__ float elu1(float x) {
    // elu(x)+1 : x>0 -> x+1 ; x<=0 -> exp(x)
    return x > 0.f ? x + 1.f : __expf(x);
}

__device__ __forceinline__ v16h cat8(v8h lo, v8h hi) {
    return __builtin_shufflevector(lo, hi, 0,1,2,3,4,5,6,7,8,9,10,11,12,13,14,15);
}

// ---------------------------------------------------------------- pass 0
__global__ void la_zero(float* __restrict__ p, int count) {
    int i = blockIdx.x * blockDim.x + threadIdx.x;
    if (i < count) p[i] = 0.f;
}

// ---------------------------------------------------------------- pass 1
// KV[n,h,d,v] += sum_s elu1(K[n,s,h,d]) * V[n,s,h,v];  K_sum[n,h,d] += sum_s elu1(K)
#define P1_STRIDE 40   // halves per LDS row (32 data + 8 pad), 80B = 16B-aligned

__global__ __launch_bounds__(256) void la_pass1(
    const float* __restrict__ Kg, const float* __restrict__ Vg,
    float* __restrict__ ws_kv, float* __restrict__ ws_ksum)
{
    __shared__ _Float16 Kt[64 * P1_STRIDE];   // Kt[d][s_local], f16 features
    __shared__ _Float16 Vt[64 * P1_STRIDE];   // Vt[v][s_local]

    const int chunk = blockIdx.x;
    const int h     = blockIdx.y;
    const int n     = blockIdx.z;
    const int tid   = threadIdx.x;
    const int lane  = tid & 31;
    const int wave  = tid >> 5;
    const int g     = lane >> 4;     // half-wave group
    const int ml    = lane & 15;
    const int s_base = chunk * CH;

    // 16 subtiles (4 d-tiles x 4 v-tiles) over 8 waves -> 2 tiles each
    const int dt  = wave >> 1;
    const int vt0 = (wave & 1) * 2;

    v8f c0 = {}; v8f c1 = {};
    float ksum_acc = 0.f;

    const size_t rs = (size_t)H_ * D_;
    const float* Kbase = Kg + ((size_t)n * S_ * H_ + h) * D_;
    const float* Vbase = Vg + ((size_t)n * S_ * H_ + h) * D_;

    for (int sc = 0; sc < CH; sc += 32) {
        __syncthreads();   // previous iteration's LDS reads complete
        // stage 32x64 K and V tiles transposed into LDS as f16 (2 quads/thread)
        #pragma unroll
        for (int i = 0; i < 2; ++i) {
            int q  = tid + i * 256;      // 0..511
            int sl = q >> 4;             // 0..31
            int dq = (q & 15) * 4;       // 0..60
            v4f kf = *(const v4f*)(Kbase + (size_t)(s_base + sc + sl) * rs + dq);
            v4f vf = *(const v4f*)(Vbase + (size_t)(s_base + sc + sl) * rs + dq);
            #pragma unroll
            for (int j = 0; j < 4; ++j) {
                Kt[(dq + j) * P1_STRIDE + sl] = (_Float16)elu1(kf[j]);
                Vt[(dq + j) * P1_STRIDE + sl] = (_Float16)vf[j];
            }
        }
        __syncthreads();

        // K_sum partial (f32 accumulation)
        if (tid < 64) {
            #pragma unroll
            for (int j = 0; j < 4; ++j) {
                v8h kk = *(const v8h*)&Kt[tid * P1_STRIDE + j * 8];
                #pragma unroll
                for (int e = 0; e < 8; ++e) ksum_acc += (float)kk[e];
            }
        }

        // A fragment: A[m][k] = Kt[dt*16+m][s], contiguous in s per layout
        const _Float16* arow = &Kt[(dt * 16 + ml) * P1_STRIDE];
        v16h a = cat8(*(const v8h*)(arow + 8 * g),
                      *(const v8h*)(arow + 16 + 8 * g));
        // B fragments: B[k][n] = Vt[vt*16+n][s], K ranges 8g..8g+7 and 16+8g..16+8g+7 (same as A)
        const _Float16* br0 = &Vt[((vt0    ) * 16 + ml) * P1_STRIDE];
        const _Float16* br1 = &Vt[((vt0 + 1) * 16 + ml) * P1_STRIDE];
        v16h b0 = cat8(*(const v8h*)(br0 + 8 * g), *(const v8h*)(br0 + 16 + 8 * g));
        v16h b1 = cat8(*(const v8h*)(br1 + 8 * g), *(const v8h*)(br1 + 16 + 8 * g));
        c0 = __builtin_amdgcn_wmma_f32_16x16x32_f16(false, a, false, b0, (short)0, c0, false, false);
        c1 = __builtin_amdgcn_wmma_f32_16x16x32_f16(false, a, false, b1, (short)0, c1, false, false);
    }

    const int nh = n * H_ + h;
    if (tid < 64) atomicAdd(&ws_ksum[nh * 64 + tid], ksum_acc);
    float* kvb = ws_kv + (size_t)nh * 64 * 64;
    #pragma unroll
    for (int r = 0; r < 8; ++r) {
        int d  = dt * 16 + r + 8 * g;     // C layout: VGPR r -> M=r (+8 for hi half)
        int v0 = vt0 * 16 + ml;
        atomicAdd(&kvb[d * 64 + v0],      c0[r]);
        atomicAdd(&kvb[d * 64 + v0 + 16], c1[r]);
    }
}

// ---------------------------------------------------------------- pass 2
// out[n,l,h,v] = (elu1(Q[n,l,h,:]) @ KV[n,h,:,:]) * Z[l],  Z = 1/(Q'.K_sum + eps)
#define QS_STRIDE 72   // 64 + 8 pad halves, 144B row (16B multiple)
#define KV_STRIDE 72

__global__ __launch_bounds__(256) void la_pass2(
    const float* __restrict__ Qg, const float* __restrict__ ws_kv,
    const float* __restrict__ ws_ksum, float* __restrict__ Og)
{
    __shared__ _Float16 Qs [LTILE * QS_STRIDE];  // Qs[l_local][d], f16 features
    __shared__ _Float16 KVt[64 * KV_STRIDE];     // KVt[v][d], f16
    __shared__ float Zs[LTILE];
    __shared__ float Ks_s[64];

    const int lt   = blockIdx.x;
    const int h    = blockIdx.y;
    const int n    = blockIdx.z;
    const int tid  = threadIdx.x;
    const int lane = tid & 31;
    const int wave = tid >> 5;
    const int g    = lane >> 4;
    const int ml   = lane & 15;
    const int l0   = lt * LTILE;
    const int nh   = n * H_ + h;
    const size_t rs = (size_t)H_ * D_;
    const float* Qbase = Qg + ((size_t)n * L_ * H_ + h) * D_;

    // stage Q' tile 128x64 (8 quads/thread), row-major in LDS
    #pragma unroll
    for (int i = 0; i < 8; ++i) {
        int q   = tid + i * 256;    // 0..2047
        int row = q >> 4;
        int dq  = (q & 15) * 4;
        v4f qf = *(const v4f*)(Qbase + (size_t)(l0 + row) * rs + dq);
        v4h qh;
        #pragma unroll
        for (int j = 0; j < 4; ++j) qh[j] = (_Float16)elu1(qf[j]);
        *(v4h*)&Qs[row * QS_STRIDE + dq] = qh;
    }
    // stage KV transposed 64x64 (4 quads/thread)
    const float* kvb = ws_kv + (size_t)nh * 4096;
    #pragma unroll
    for (int i = 0; i < 4; ++i) {
        int q  = tid + i * 256;     // 0..1023
        int d  = q >> 4;
        int vq = (q & 15) * 4;
        v4f kvf = *(const v4f*)(kvb + d * 64 + vq);
        #pragma unroll
        for (int j = 0; j < 4; ++j)
            KVt[(vq + j) * KV_STRIDE + d] = (_Float16)kvf[j];
    }
    if (tid < 64) Ks_s[tid] = ws_ksum[nh * 64 + tid];
    __syncthreads();

    // Z per row (f32)
    if (tid < LTILE) {
        float dot = 0.f;
        #pragma unroll
        for (int j = 0; j < 8; ++j) {
            v8h qh = *(const v8h*)&Qs[tid * QS_STRIDE + j * 8];
            #pragma unroll
            for (int e = 0; e < 8; ++e) dot += (float)qh[e] * Ks_s[j * 8 + e];
        }
        Zs[tid] = 1.f / (dot + EPSC);
    }
    __syncthreads();

    // wave = m-subtile (16 rows); 4 v-tiles; K-dim 64 = 2 steps of 32
    v8f c[4] = {{}, {}, {}, {}};
    #pragma unroll
    for (int ks = 0; ks < 2; ++ks) {
        const _Float16* arow = &Qs[(wave * 16 + ml) * QS_STRIDE + 32 * ks];
        v16h a = cat8(*(const v8h*)(arow + 8 * g),
                      *(const v8h*)(arow + 16 + 8 * g));
        #pragma unroll
        for (int vt = 0; vt < 4; ++vt) {
            const _Float16* brow = &KVt[(vt * 16 + ml) * KV_STRIDE + 32 * ks];
            v16h b = cat8(*(const v8h*)(brow + 8 * g), *(const v8h*)(brow + 16 + 8 * g));
            c[vt] = __builtin_amdgcn_wmma_f32_16x16x32_f16(false, a, false, b, (short)0, c[vt], false, false);
        }
    }

    // epilogue: scale by Z, coalesced f32 stores (lane index = v)
    float* obase = Og + ((size_t)n * L_ * H_ + h) * D_;
    #pragma unroll
    for (int r = 0; r < 8; ++r) {
        int lrow = wave * 16 + r + 8 * g;
        float z  = Zs[lrow];
        size_t orow = (size_t)(l0 + lrow) * rs;
        #pragma unroll
        for (int vt = 0; vt < 4; ++vt)
            obase[orow + vt * 16 + ml] = c[vt][r] * z;
    }
}

// ---------------------------------------------------------------- launch
extern "C" void kernel_launch(void* const* d_in, const int* in_sizes, int n_in,
                              void* d_out, int out_size, void* d_ws, size_t ws_size,
                              hipStream_t stream) {
    const float* Qg = (const float*)d_in[0];   // queries [N,L,H,D]
    const float* Kg = (const float*)d_in[1];   // keys    [N,S,H,D]
    const float* Vg = (const float*)d_in[2];   // values  [N,S,H,D]
    float* Og = (float*)d_out;                 // out     [N,L,H,D]

    float* ws_kv   = (float*)d_ws;                      // [N*H,64,64]
    float* ws_ksum = ws_kv + (size_t)N_ * H_ * 64 * 64; // [N*H,64]
    int zcount = N_ * H_ * 64 * 64 + N_ * H_ * 64;

    la_zero<<<dim3((zcount + 255) / 256), dim3(256), 0, stream>>>(ws_kv, zcount);
    la_pass1<<<dim3(S_ / CH, H_, N_), dim3(256), 0, stream>>>(Kg, Vg, ws_kv, ws_ksum);
    la_pass2<<<dim3(L_ / LTILE, H_, N_), dim3(256), 0, stream>>>(Qg, ws_kv, ws_ksum, Og);
}
